// LSTMCombine_50156628083031
// MI455X (gfx1250) — compile-verified, matrix operand order checked by reading, not run on hardware
//
#include <hip/hip_runtime.h>
#include <hip/hip_bf16.h>

// ---------------------------------------------------------------------------
// TFT VariableSelection (per-variable GRNs + flattened GRN + softmax select)
// B=64 T=512 E=H=160 V=16.  All matmuls via v_wmma_f32_16x16x32_bf16.
// Intermediate "transformed" stored bf16 [tok][h][v] in d_ws (~168MB assumed).
// ---------------------------------------------------------------------------

typedef __bf16 bf16;
typedef bf16  v8bf  __attribute__((ext_vector_type(8)));
typedef bf16  v16bf __attribute__((ext_vector_type(16)));
typedef float v8f   __attribute__((ext_vector_type(8)));

#define B_      64
#define T_      512
#define E_      160
#define V_      16
#define H_      160
#define NT      10          // H_/16 n-tiles
#define FLAT_   2560        // V_*H_
#define TOK_BLK 128
#define NTOK    32768       // B_*T_
#define KC2     256         // K-chunk for flattened GRN

// Load one 16x32 bf16 A/B fragment (ISA 7.12.2 16-bit layout) from a
// row-major tile in LDS: lane L<16 -> row L, K {k0..k0+7, k0+16..k0+23};
// lane L>=16 -> row L-16, K {k0+8..15, k0+24..31}. Two ds_load_b128 per lane.
static __device__ __forceinline__ v16bf load_frag(const bf16* base, int ld, int k0) {
  const int lane = threadIdx.x & 31;
  const bf16* p = base + (lane & 15) * ld + k0 + ((lane >> 4) << 3);
  v8bf lo = *(const v8bf*)p;
  v8bf hi = *(const v8bf*)(p + 16);
  v16bf r;
#pragma unroll
  for (int i = 0; i < 8; ++i) { r[i] = lo[i]; r[i + 8] = hi[i]; }
  return r;
}

static __device__ __forceinline__ v8f wmma_bf16(v16bf a, v16bf b, v8f c) {
  return __builtin_amdgcn_wmma_f32_16x16x32_bf16(false, a, false, b, (short)0, c,
                                                 false, false);
}

static __device__ __forceinline__ float elu_f(float x) {
  return x > 0.f ? x : (__expf(x) - 1.f);
}
static __device__ __forceinline__ float sigmoid_f(float x) {
  return __builtin_amdgcn_rcpf(1.f + __expf(-x));   // v_rcp_f32, no IEEE div
}

// ---------------------------------------------------------------------------
// Kernel 0: ctx[b][n] = additional_context[b] @ fl_Wc   (tiny: 64x160)
// ---------------------------------------------------------------------------
__global__ void __launch_bounds__(256) ctx_kernel(const float* __restrict__ ac,
                                                  const float* __restrict__ Wc,
                                                  float* __restrict__ ctx) {
  int idx = blockIdx.x * 256 + threadIdx.x;
  if (idx >= B_ * H_) return;
  int b = idx / H_, n = idx - b * H_;
  float s = 0.f;
  for (int k = 0; k < H_; ++k) s += ac[b * H_ + k] * Wc[k * H_ + n];
  ctx[idx] = s;
}

// ---------------------------------------------------------------------------
// Kernel 1 helpers (force-inlined with direct smem pointers so address-space
// inference keeps every tile access on the DS path, not FLAT).
// ---------------------------------------------------------------------------
static __device__ __forceinline__ void stage_wt(const float* __restrict__ Wsrc,
                                                bf16* Wt, int tid) {
  for (int idx = tid; idx < H_ * H_; idx += 256) {
    int k = idx / H_, n = idx - k * H_;
    Wt[n * H_ + k] = (bf16)Wsrc[idx];
  }
}

// ACT: 0 = identity, 1 = ELU, 2 = sigmoid
template <int ACT>
static __device__ __forceinline__ void layer_mm(const bf16* Awave, const bf16* Wt,
                                                const float* __restrict__ bias,
                                                bf16* dstWave, int col, int rbase) {
  for (int nt = 0; nt < NT; ++nt) {
    v8f acc;
    float bvv = bias[nt * 16 + col];
#pragma unroll
    for (int i = 0; i < 8; ++i) acc[i] = bvv;
#pragma unroll
    for (int kc = 0; kc < 5; ++kc) {
      v16bf a = load_frag(Awave, H_, kc * 32);
      v16bf b = load_frag(Wt + nt * 16 * H_, H_, kc * 32);
      acc = wmma_bf16(a, b, acc);
    }
#pragma unroll
    for (int r = 0; r < 8; ++r) {
      float t = acc[r];
      if (ACT == 1) t = elu_f(t);
      if (ACT == 2) t = sigmoid_f(t);
      dstWave[(rbase + r) * H_ + nt * 16 + col] = (bf16)t;
    }
  }
}

// ---------------------------------------------------------------------------
// Kernel 1: per-variable GRN. Block = (token tile of 128, variable v).
// 8 waves x 16 tokens each. Weights transposed->bf16 staged in LDS per layer.
// ---------------------------------------------------------------------------
__global__ void __launch_bounds__(256) grn_pervar_kernel(
    const float* __restrict__ emb,
    const float* __restrict__ W1, const float* __restrict__ b1,
    const float* __restrict__ W2, const float* __restrict__ b2,
    const float* __restrict__ Wg, const float* __restrict__ bg,
    const float* __restrict__ Wv, const float* __restrict__ bv,
    const float* __restrict__ lns, const float* __restrict__ lnb,
    bf16* __restrict__ trbuf) {
  extern __shared__ char smem[];
  bf16*  xA = (bf16*)smem;            // [128][160] bf16 input (also skip)
  bf16*  hA = xA + TOK_BLK * H_;      // ping
  bf16*  hB = hA + TOK_BLK * H_;      // pong
  bf16*  Wt = hB + TOK_BLK * H_;      // [160][160] transposed weights bf16
  float* yv = (float*)(Wt + H_ * H_); // [128][160] pre-LN fp32

  const int v     = blockIdx.x & (V_ - 1);
  const int tile  = blockIdx.x >> 4;
  const int tok0  = tile * TOK_BLK;
  const int tid   = threadIdx.x;
  const int lane  = tid & 31;
  const int wave  = tid >> 5;
  const int col   = lane & 15;
  const int rbase = (lane >> 4) << 3;
  const int woff  = wave * 16 * H_;

  // Stage x_v (strided gather over v; lines shared across the 16 v-blocks).
  for (int idx = tid; idx < TOK_BLK * E_; idx += 256) {
    int tok = idx / E_, e = idx - tok * E_;
    xA[idx] = (bf16)emb[((long)(tok0 + tok) * E_ + e) * V_ + v];
  }

  // ---- layer 1: h1 = elu(x @ W1 + b1) ----
  __syncthreads();
  stage_wt(W1 + (long)v * E_ * H_, Wt, tid);
  __syncthreads();
  layer_mm<1>(xA + woff, Wt, b1 + v * H_, hA + woff, col, rbase);

  // ---- layer 2: h2 = h1 @ W2 + b2 ----
  __syncthreads();
  stage_wt(W2 + (long)v * H_ * H_, Wt, tid);
  __syncthreads();
  layer_mm<0>(hA + woff, Wt, b2 + v * H_, hB + woff, col, rbase);

  // ---- layer 3: gate = sigmoid(h2 @ Wg + bg) ----
  __syncthreads();
  stage_wt(Wg + (long)v * H_ * H_, Wt, tid);
  __syncthreads();
  layer_mm<2>(hB + woff, Wt, bg + v * H_, hA + woff, col, rbase);

  // ---- layer 4: val = h2 @ Wv + bv ; y = x + gate*val ----
  __syncthreads();
  stage_wt(Wv + (long)v * H_ * H_, Wt, tid);
  __syncthreads();
  {
    const float* bias = bv + v * H_;
    for (int nt = 0; nt < NT; ++nt) {
      v8f acc;
      float bvv = bias[nt * 16 + col];
#pragma unroll
      for (int i = 0; i < 8; ++i) acc[i] = bvv;
#pragma unroll
      for (int kc = 0; kc < 5; ++kc) {
        v16bf a = load_frag(hB + woff, H_, kc * 32);
        v16bf b = load_frag(Wt + nt * 16 * H_, H_, kc * 32);
        acc = wmma_bf16(a, b, acc);
      }
#pragma unroll
      for (int r = 0; r < 8; ++r) {
        int row = wave * 16 + rbase + r;
        int n   = nt * 16 + col;
        float g = (float)hA[row * H_ + n];
        float x = (float)xA[row * H_ + n];
        yv[row * H_ + n] = x + g * acc[r];
      }
    }
  }
  __syncthreads();

  // LayerNorm over H, write transformed bf16 at [tok][h][v].
  if (tid < TOK_BLK) {
    const float* y = yv + tid * H_;
    float m = 0.f;
    for (int h = 0; h < H_; ++h) m += y[h];
    m *= (1.f / H_);
    float var = 0.f;
    for (int h = 0; h < H_; ++h) { float d = y[h] - m; var += d * d; }
    var *= (1.f / H_);
    float rs = __builtin_amdgcn_rsqf(var + 1e-5f);
    long base = (long)(tok0 + tid) * FLAT_ + v;
    for (int h = 0; h < H_; ++h) {
      float o = (y[h] - m) * rs * lns[v * H_ + h] + lnb[v * H_ + h];
      trbuf[base + (long)h * V_] = (bf16)o;
    }
  }
}

// ---------------------------------------------------------------------------
// Kernel 2: flattened GRN + softmax + weighted sum. Block = 128 tokens.
// ---------------------------------------------------------------------------
__global__ void __launch_bounds__(256) flat_grn_kernel(
    const bf16* __restrict__ trbuf, const float* __restrict__ ctxbuf,
    const float* __restrict__ Wskip, const float* __restrict__ bskip,
    const float* __restrict__ W1, const float* __restrict__ b1,
    const float* __restrict__ W2, const float* __restrict__ b2,
    const float* __restrict__ Wg, const float* __restrict__ bg,
    const float* __restrict__ Wv, const float* __restrict__ bv,
    const float* __restrict__ lns, const float* __restrict__ lnb,
    float* __restrict__ outTC, float* __restrict__ outSW,
    float* __restrict__ outSG) {
  extern __shared__ char smem[];
  bf16*  Abuf = (bf16*)smem;              // [128][256]; reused as h2/h3 [128][160]
  bf16*  Bbuf = Abuf + TOK_BLK * KC2;     // [160][256]; reused as W2^T [160][160]
  bf16*  Sbuf = Bbuf + H_ * KC2;          // [16][256];  reused as Wg^T+Wv^T
  float* yws  = (float*)(Sbuf + 2 * V_ * H_); // 8 waves x 16x16 pre-LN

  const int tok0  = blockIdx.x * TOK_BLK;
  const int bidx  = tok0 / T_;            // 512 % 128 == 0: one batch per block
  const int tid   = threadIdx.x;
  const int lane  = tid & 31;
  const int wave  = tid >> 5;
  const int col   = lane & 15;
  const int rbase = (lane >> 4) << 3;
  const int wtok  = tok0 + wave * 16;

  v8f accH[NT], accS;
  {
    float s = bskip[col];
#pragma unroll
    for (int i = 0; i < 8; ++i) accS[i] = s;
#pragma unroll
    for (int nt = 0; nt < NT; ++nt) {
      float t = b1[nt * 16 + col] + ctxbuf[bidx * H_ + nt * 16 + col];
#pragma unroll
      for (int i = 0; i < 8; ++i) accH[nt][i] = t;
    }
  }

  // --- K loop over flat dim (2560) in 256-chunks: fl_W1 and fl_Wskip ---
  for (int kc = 0; kc < FLAT_ / KC2; ++kc) {
    __syncthreads();
    // A tile: contiguous bf16 global -> LDS via CDNA5 async DMA (16B/lane).
    for (int c = tid; c < TOK_BLK * KC2 / 8; c += 256) {
      int tok = (c * 8) / KC2;
      int off = (c * 8) - tok * KC2;
      const bf16* src = trbuf + (long)(tok0 + tok) * FLAT_ + kc * KC2 + off;
      bf16*       dst = Abuf + tok * KC2 + off;
      unsigned      lds_off = (unsigned)(unsigned long)dst;  // low 32b = LDS addr
      unsigned long gaddr   = (unsigned long)src;
      asm volatile("global_load_async_to_lds_b128 %0, %1, off"
                   :: "v"(lds_off), "v"(gaddr) : "memory");
    }
    for (int idx = tid; idx < H_ * KC2; idx += 256) {      // W1^T chunk
      int kk = idx / H_, n = idx - kk * H_;
      Bbuf[n * KC2 + kk] = (bf16)W1[(long)(kc * KC2 + kk) * H_ + n];
    }
    for (int idx = tid; idx < V_ * KC2; idx += 256) {      // Wskip^T chunk
      int kk = idx / V_, j = idx - kk * V_;
      Sbuf[j * KC2 + kk] = (bf16)Wskip[(long)(kc * KC2 + kk) * V_ + j];
    }
    asm volatile("s_wait_asynccnt 0x0" ::: "memory");
    __syncthreads();
    const bf16* A = Abuf + wave * 16 * KC2;
    for (int kcc = 0; kcc < KC2 / 32; ++kcc) {
      v16bf a = load_frag(A, KC2, kcc * 32);
#pragma unroll
      for (int nt = 0; nt < NT; ++nt) {
        v16bf b = load_frag(Bbuf + nt * 16 * KC2, KC2, kcc * 32);
        accH[nt] = wmma_bf16(a, b, accH[nt]);
      }
      v16bf bs = load_frag(Sbuf, KC2, kcc * 32);
      accS = wmma_bf16(a, bs, accS);
    }
  }
  __syncthreads();

  // h2 = elu(.) -> LDS [128][160]; stage W2^T, Wg^T, Wv^T.
  bf16* h2A = Abuf;
#pragma unroll
  for (int nt = 0; nt < NT; ++nt)
#pragma unroll
    for (int r = 0; r < 8; ++r)
      h2A[(wave * 16 + rbase + r) * H_ + nt * 16 + col] = (bf16)elu_f(accH[nt][r]);
  bf16* W2t = Bbuf;
  for (int idx = tid; idx < H_ * H_; idx += 256) {
    int k = idx / H_, n = idx - k * H_;
    W2t[n * H_ + k] = (bf16)W2[idx];
  }
  for (int idx = tid; idx < V_ * H_; idx += 256) {         // fl_Wg/fl_Wv [160][16]
    int k = idx / V_, j = idx - k * V_;
    Sbuf[j * H_ + k]           = (bf16)Wg[idx];
    Sbuf[V_ * H_ + j * H_ + k] = (bf16)Wv[idx];
  }
  __syncthreads();

  // h3 = h2 @ W2 + b2
  v8f accH2[NT];
#pragma unroll
  for (int nt = 0; nt < NT; ++nt) {
    float t = b2[nt * 16 + col];
#pragma unroll
    for (int i = 0; i < 8; ++i) accH2[nt][i] = t;
  }
  const bf16* A2 = h2A + wave * 16 * H_;
  for (int kc = 0; kc < 5; ++kc) {
    v16bf a = load_frag(A2, H_, kc * 32);
#pragma unroll
    for (int nt = 0; nt < NT; ++nt) {
      v16bf b = load_frag(W2t + nt * 16 * H_, H_, kc * 32);
      accH2[nt] = wmma_bf16(a, b, accH2[nt]);
    }
  }
  // h3 back to LDS (own rows only; same-wave DS ordering is sufficient)
#pragma unroll
  for (int nt = 0; nt < NT; ++nt)
#pragma unroll
    for (int r = 0; r < 8; ++r)
      h2A[(wave * 16 + rbase + r) * H_ + nt * 16 + col] = (bf16)accH2[nt][r];

  // gate / value heads (N = 16)
  v8f accG, accV;
  {
    float g = bg[col], vv = bv[col];
#pragma unroll
    for (int i = 0; i < 8; ++i) { accG[i] = g; accV[i] = vv; }
  }
  for (int kc = 0; kc < 5; ++kc) {
    v16bf a  = load_frag(A2, H_, kc * 32);
    v16bf bG = load_frag(Sbuf, H_, kc * 32);
    v16bf bV = load_frag(Sbuf + V_ * H_, H_, kc * 32);
    accG = wmma_bf16(a, bG, accG);
    accV = wmma_bf16(a, bV, accV);
  }

  // static_gate out; y = skip + gate*val into yws for row-wise LN/softmax.
#pragma unroll
  for (int r = 0; r < 8; ++r) {
    float g = sigmoid_f(accG[r]);
    outSG[(long)(wtok + rbase + r) * V_ + col] = g;
    yws[wave * 256 + (rbase + r) * 16 + col] = accS[r] + g * accV[r];
  }
  __syncthreads();

  // LN over V, softmax, then temporal_ctx = sum_v transformed*w. Both lane
  // halves redo the small LN/softmax for row=lane&15; split the 160 h's.
  {
    int row = lane & 15;
    int token = wtok + row;
    const float* yr = yws + wave * 256 + row * 16;
    float m = 0.f;
#pragma unroll
    for (int j = 0; j < 16; ++j) m += yr[j];
    m *= (1.f / 16.f);
    float var = 0.f;
#pragma unroll
    for (int j = 0; j < 16; ++j) { float d = yr[j] - m; var += d * d; }
    var *= (1.f / 16.f);
    float rs = __builtin_amdgcn_rsqf(var + 1e-5f);
    float w[16], mx = -1e30f;
#pragma unroll
    for (int j = 0; j < 16; ++j) {
      float l = (yr[j] - m) * rs * lns[j] + lnb[j];
      w[j] = l; mx = fmaxf(mx, l);
    }
    float ssum = 0.f;
#pragma unroll
    for (int j = 0; j < 16; ++j) { w[j] = __expf(w[j] - mx); ssum += w[j]; }
    float inv = __builtin_amdgcn_rcpf(ssum);
#pragma unroll
    for (int j = 0; j < 16; ++j) w[j] *= inv;
    if (lane < 16) {
#pragma unroll
      for (int j = 0; j < 16; ++j) outSW[(long)token * V_ + j] = w[j];
    }
    const bf16* tr = trbuf + (long)token * FLAT_;
    int h0 = (lane >> 4) * 80;
    for (int h = h0; h < h0 + 80; ++h) {
      v8bf t0 = *(const v8bf*)(tr + h * V_);
      v8bf t1 = *(const v8bf*)(tr + h * V_ + 8);
      float acc = 0.f;
#pragma unroll
      for (int j = 0; j < 8; ++j)
        acc += (float)t0[j] * w[j] + (float)t1[j] * w[j + 8];
      outTC[(long)token * H_ + h] = acc;
    }
  }
}

// ---------------------------------------------------------------------------
extern "C" void kernel_launch(void* const* d_in, const int* in_sizes, int n_in,
                              void* d_out, int out_size, void* d_ws,
                              size_t ws_size, hipStream_t stream) {
  const float* emb    = (const float*)d_in[0];
  const float* ac     = (const float*)d_in[1];
  const float* svW1   = (const float*)d_in[2];
  const float* svb1   = (const float*)d_in[3];
  const float* svW2   = (const float*)d_in[4];
  const float* svb2   = (const float*)d_in[5];
  const float* svWg   = (const float*)d_in[6];
  const float* svbg   = (const float*)d_in[7];
  const float* svWv   = (const float*)d_in[8];
  const float* svbv   = (const float*)d_in[9];
  const float* svlns  = (const float*)d_in[10];
  const float* svlnb  = (const float*)d_in[11];
  const float* flWskip= (const float*)d_in[12];
  const float* flbskip= (const float*)d_in[13];
  const float* flW1   = (const float*)d_in[14];
  const float* flb1   = (const float*)d_in[15];
  const float* flWc   = (const float*)d_in[16];
  const float* flW2   = (const float*)d_in[17];
  const float* flb2   = (const float*)d_in[18];
  const float* flWg   = (const float*)d_in[19];
  const float* flbg   = (const float*)d_in[20];
  const float* flWv   = (const float*)d_in[21];
  const float* flbv   = (const float*)d_in[22];
  const float* fllns  = (const float*)d_in[23];
  const float* fllnb  = (const float*)d_in[24];

  // workspace: ctx (64x160 f32) at 0; transformed bf16 [NTOK][H][V] at 64KB.
  float* ctxbuf = (float*)d_ws;
  bf16*  trbuf  = (bf16*)((char*)d_ws + 65536);

  float* outTC = (float*)d_out;
  float* outSW = outTC + (long)NTOK * H_;
  float* outSG = outSW + (long)NTOK * V_;

  size_t smem1 = (size_t)(3 * TOK_BLK * H_ + H_ * H_) * sizeof(bf16) +
                 (size_t)TOK_BLK * H_ * sizeof(float);           // ~256 KB
  size_t smem2 = (size_t)(TOK_BLK * KC2 + H_ * KC2 + 2 * V_ * H_) * sizeof(bf16) +
                 (size_t)8 * 256 * sizeof(float);                // ~166 KB

  (void)hipFuncSetAttribute(reinterpret_cast<const void*>(grn_pervar_kernel),
                            hipFuncAttributeMaxDynamicSharedMemorySize, (int)smem1);
  (void)hipFuncSetAttribute(reinterpret_cast<const void*>(flat_grn_kernel),
                            hipFuncAttributeMaxDynamicSharedMemorySize, (int)smem2);

  ctx_kernel<<<(B_ * H_ + 255) / 256, 256, 0, stream>>>(ac, flWc, ctxbuf);
  grn_pervar_kernel<<<V_ * (NTOK / TOK_BLK), 256, smem1, stream>>>(
      emb, svW1, svb1, svW2, svb2, svWg, svbg, svWv, svbv, svlns, svlnb, trbuf);
  flat_grn_kernel<<<NTOK / TOK_BLK, 256, smem2, stream>>>(
      trbuf, ctxbuf, flWskip, flbskip, flW1, flb1, flW2, flb2, flWg, flbg,
      flWv, flbv, fllns, fllnb, outTC, outSW, outSG);
}
